// GraphSAGE_44444321579079
// MI455X (gfx1250) — matrix operand order repaired and checked by reading, not compiled, not run on hardware
//
#include <hip/hip_runtime.h>

// GraphSAGE (LSTM aggregator) for MI455X / gfx1250, wave32 WMMA + TDM.
// N=50000, DEG=16, FIN=HID=128, OUT=64, 3 layers.
//
// v2 changes vs v1:
//  - WMMA chains start from inline-zero C; bias folded into pointwise stage
//    (bias staged in LDS) -> removes per-timestep acc-init mov storm.
//  - A-tiles (x_t and h) preloaded into registers per timestep; 4 independent
//    accumulation chains retained for XDL ILP.
//  - Wih/Whh staged into LDS via TENSOR_LOAD_TO_LDS (TDM) + s_wait_tensorcnt,
//    issued by wave 0; cooperative-load fallback if builtin absent.

typedef __attribute__((ext_vector_type(16))) _Float16 v16h;
typedef __attribute__((ext_vector_type(8)))  float    v8f;
typedef __attribute__((ext_vector_type(4)))  unsigned int u32x4;
typedef __attribute__((ext_vector_type(8)))  int      i32x8;
typedef __attribute__((ext_vector_type(4)))  int      i32x4;

#define WMMA_F16(a, b, c) \
  __builtin_amdgcn_wmma_f32_16x16x32_f16(false, (a), false, (b), (short)0, (c), false, false)

#if __has_builtin(__builtin_amdgcn_tensor_load_to_lds)
#define HAVE_TDM 1
#else
#define HAVE_TDM 0
#endif

static constexpr int    DEG = 16;
static constexpr int    FI  = 128;
static constexpr size_t SAGE_SMEM =
    (size_t)512 * 128 * 2     // wih_s  f16
  + (size_t)512 * 128 * 2     // whh_s  f16
  + (size_t)16 * 512 * 4      // gate_s f32
  + (size_t)2 * 16 * 128 * 2  // xt_s   f16 (double buffer)
  + (size_t)16 * 128 * 2      // h_s    f16
  + (size_t)512 * 4;          // bias_s f32
// = 309248 bytes, fits CDNA5's 320KB-per-WG LDS.

// ---------- small prep kernels ----------
__global__ void k_f32_to_f16(const float* __restrict__ s, _Float16* __restrict__ d, int n) {
  int i  = blockIdx.x * blockDim.x + threadIdx.x;
  int st = gridDim.x * blockDim.x;
  for (; i < n; i += st) d[i] = (_Float16)s[i];
}

__global__ void k_bias_sum(const float* __restrict__ a, const float* __restrict__ b,
                           float* __restrict__ o, int n) {
  int i = blockIdx.x * blockDim.x + threadIdx.x;
  if (i < n) o[i] = a[i] + b[i];
}

// ---------- TDM: 2D tile (f16) global -> LDS ----------
#if HAVE_TDM
__device__ __forceinline__ void tdm_load_2d_f16(unsigned lds_off, const void* gptr,
                                                unsigned dim0, unsigned dim1) {
  unsigned long long ga = (unsigned long long)(uintptr_t)gptr;
  u32x4 g0;
  g0.x = 1u;                                  // count=1, is_restore=0, no gather
  g0.y = lds_off;                             // lds_addr
  g0.z = (unsigned)ga;                        // global_addr[31:0]
  g0.w = (unsigned)((ga >> 32) & 0x01FFFFFFu) | (2u << 30);  // addr[56:32] | type=2
  i32x8 g1;
  g1[0] = 0x00010000;                         // wg_mask=0, data_size=1 (2B)
  g1[1] = (int)((dim0 & 0xFFFFu) << 16);      // tensor_dim0[15:0]
  g1[2] = (int)((dim0 >> 16) | ((dim1 & 0xFFFFu) << 16));  // dim0[31:16], dim1[15:0]
  g1[3] = (int)((dim1 >> 16) | ((dim0 & 0xFFFFu) << 16));  // dim1[31:16], tile_dim0
  g1[4] = (int)(dim1 & 0xFFFFu);              // tile_dim1, tile_dim2=0
  g1[5] = (int)dim0;                          // tensor_dim0_stride[31:0]
  g1[6] = 0;                                  // stride hi, dim1_stride lo
  g1[7] = 0;
  i32x4 z4 = {0, 0, 0, 0};
#if __clang_major__ >= 23
  i32x8 z8 = {0, 0, 0, 0, 0, 0, 0, 0};
  __builtin_amdgcn_tensor_load_to_lds(g0, g1, z4, z4, z8, 0);
#else
  __builtin_amdgcn_tensor_load_to_lds(g0, g1, z4, z4, 0);
#endif
}
#endif

// ---------- WMMA operand loaders (ISA 7.12.2 layouts, wave32) ----------
__device__ __forceinline__ v16h load_a_tile(const _Float16* __restrict__ base,
                                            int lane, int kt) {
  int m = lane & 15, half = lane >> 4;
  const _Float16* row = base + m * 128 + kt * 32 + half * 8;
  union { uint4 q[2]; v16h v; } u;
  u.q[0] = *(const uint4*)(row);
  u.q[1] = *(const uint4*)(row + 16);
  return u.v;
}

__device__ __forceinline__ v16h load_b_tile(const _Float16* __restrict__ w,
                                            int lane, int n0, int kt) {
  const _Float16* row = w + (size_t)(n0 + (lane & 15)) * 128 + kt * 32 + (lane >> 4) * 8;
  union { uint4 q[2]; v16h v; } u;
  u.q[0] = *(const uint4*)(row);
  u.q[1] = *(const uint4*)(row + 16);
  return u.v;
}

__device__ __forceinline__ float fsigmoid(float x) { return 1.0f / (1.0f + __expf(-x)); }
__device__ __forceinline__ float ftanh(float x)    { return 2.0f / (1.0f + __expf(-2.0f * x)) - 1.0f; }

// ---------- main layer kernel ----------
// One block = 16 dst nodes, 256 threads = 8 waves.
// Wave w: gate g = w>>1, N-half nh = w&1 -> N-tiles (nh*4+j)*16, j=0..3.
template <int FO, bool RELU, bool FINAL>
__global__ __launch_bounds__(256)
void sage_layer(const _Float16* __restrict__ feat_in,  // [N,128] f16
                const int*      __restrict__ src,       // [N,16]
                const _Float16* __restrict__ wih,       // [512,128] f16
                const _Float16* __restrict__ whh,       // [512,128] f16
                const float*    __restrict__ bias,      // [512] (b_ih + b_hh)
                const _Float16* __restrict__ wl,        // [FO,128] f16
                const float*    __restrict__ bl,        // [FO]
                const _Float16* __restrict__ wr,        // [FO,128] f16
                _Float16*       __restrict__ feat_out,  // [N,FO] f16 (if !FINAL)
                float*          __restrict__ out_f32,   // [N,FO] f32 (if FINAL)
                int n_nodes) {
  extern __shared__ char lds_raw[];
  _Float16* wih_s  = (_Float16*)lds_raw;               // 512*128
  _Float16* whh_s  = wih_s + 512 * 128;                // 512*128
  float*    gate_s = (float*)(whh_s + 512 * 128);      // [16][512]
  _Float16* xt_s   = (_Float16*)(gate_s + 16 * 512);   // 2 x [16][128]
  _Float16* h_s    = xt_s + 2 * 16 * 128;              // [16][128]
  float*    bias_s = (float*)(h_s + 16 * 128);         // [512]

  const int tid  = threadIdx.x;
  const int lane = tid & 31;
  const int wave = tid >> 5;
  const int g    = wave >> 1;
  const int nh   = wave & 1;
  const int base = blockIdx.x * 16;

  // --- stage Wih/Whh into LDS ---
#if HAVE_TDM
  if (wave == 0) {
    tdm_load_2d_f16((unsigned)(uintptr_t)wih_s, wih, 128u, 512u);
    tdm_load_2d_f16((unsigned)(uintptr_t)whh_s, whh, 128u, 512u);
    __builtin_amdgcn_s_wait_tensorcnt(0);
  }
#else
  {
    const uint4* wg = (const uint4*)wih;
    const uint4* hg = (const uint4*)whh;
    uint4* ws = (uint4*)wih_s;
    uint4* hs = (uint4*)whh_s;
    for (int i = tid; i < 512 * 128 / 8; i += 256) { ws[i] = wg[i]; hs[i] = hg[i]; }
  }
#endif
  // bias -> LDS
  for (int i = tid; i < 512; i += 256) bias_s[i] = bias[i];

  // --- prologue: gather x_0 into xt buffer 0, zero h ---
  const int nd  = tid >> 4;        // node 0..15
  const int seg = tid & 15;        // 8-f16 segment of the 128-wide row
  {
    int node = base + nd; if (node >= n_nodes) node = n_nodes - 1;
    int idx  = src[node * DEG + 0];
    ((uint4*)xt_s)[nd * 16 + seg] =
        ((const uint4*)(feat_in + (size_t)idx * 128))[seg];
    uint4 z; z.x = z.y = z.z = z.w = 0u;
    ((uint4*)h_s)[tid] = z;
  }
  __syncthreads();

  const int c0 = (tid & 15) * 8;   // pointwise cols c0..c0+7 of node nd
  float cst[8];
#pragma unroll
  for (int e = 0; e < 8; ++e) cst[e] = 0.0f;

  const int gcol = g * 128 + nh * 64;   // first output col of this wave

  // --- 16 LSTM timesteps ---
  for (int t = 0; t < 16; ++t) {
    // issue next-step gather early (latency hidden behind WMMA block)
    uint4 pf;
    if (t < 15) {
      int node = base + nd; if (node >= n_nodes) node = n_nodes - 1;
      int idx  = src[node * DEG + (t + 1)];
      pf = ((const uint4*)(feat_in + (size_t)idx * 128))[seg];
    }

    // preload all A tiles for this timestep
    const _Float16* xbuf = xt_s + (t & 1) * (16 * 128);
    v16h axr[4], ahr[4];
#pragma unroll
    for (int kt = 0; kt < 4; ++kt) {
      axr[kt] = load_a_tile(xbuf, lane, kt);
      ahr[kt] = load_a_tile(h_s,  lane, kt);
    }

    // 4 independent accumulation chains; first WMMA uses inline-zero C.
    v8f zero8 = {};
    v8f acc[4];
#pragma unroll
    for (int j = 0; j < 4; ++j) {
      int n0 = gcol + j * 16;
      v16h bi = load_b_tile(wih_s, lane, n0, 0);
      acc[j] = WMMA_F16(axr[0], bi, zero8);
      v16h bh = load_b_tile(whh_s, lane, n0, 0);
      acc[j] = WMMA_F16(ahr[0], bh, acc[j]);
    }
#pragma unroll
    for (int kt = 1; kt < 4; ++kt) {
#pragma unroll
      for (int j = 0; j < 4; ++j) {
        int n0 = gcol + j * 16;
        v16h bi = load_b_tile(wih_s, lane, n0, kt);
        acc[j] = WMMA_F16(axr[kt], bi, acc[j]);
        v16h bh = load_b_tile(whh_s, lane, n0, kt);
        acc[j] = WMMA_F16(ahr[kt], bh, acc[j]);
      }
    }

    // commit prefetched gather for t+1
    if (t < 15)
      ((uint4*)(xt_s + ((t + 1) & 1) * (16 * 128)))[nd * 16 + seg] = pf;

    // spill gates to LDS f32 scratch (C layout: lane L vgpr r -> M=(L/16)*8+r, N=L%16)
    {
      const int mb = (lane >> 4) * 8;
#pragma unroll
      for (int j = 0; j < 4; ++j) {
        int col = gcol + j * 16 + (lane & 15);
#pragma unroll
        for (int r = 0; r < 8; ++r)
          gate_s[(mb + r) * 512 + col] = acc[j][r];
      }
    }
    __syncthreads();

    // pointwise LSTM update (bias added here); gate order i,f,g,o
    {
      const float* gr = gate_s + nd * 512 + c0;
      const float* bs = bias_s + c0;
#pragma unroll
      for (int e = 0; e < 8; ++e) {
        float vi = gr[0 * 128 + e] + bs[0 * 128 + e];
        float vf = gr[1 * 128 + e] + bs[1 * 128 + e];
        float vg = gr[2 * 128 + e] + bs[2 * 128 + e];
        float vo = gr[3 * 128 + e] + bs[3 * 128 + e];
        float c  = fsigmoid(vf) * cst[e] + fsigmoid(vi) * ftanh(vg);
        cst[e]   = c;
        h_s[nd * 128 + c0 + e] = (_Float16)(fsigmoid(vo) * ftanh(c));
      }
    }
    __syncthreads();
  }

  // --- output: out = h @ Wl^T + bl + x_self @ Wr^T ---
  {
    int node = base + nd; if (node >= n_nodes) node = n_nodes - 1;
    ((uint4*)xt_s)[nd * 16 + seg] =
        ((const uint4*)(feat_in + (size_t)node * 128))[seg];
  }
  __syncthreads();

  constexpr int NT = FO / 16;      // 8 (hidden) or 4 (final)
  if (wave < NT) {
    const int nt = wave;
    const float blv = bl[nt * 16 + (lane & 15)];
    v8f zero8 = {};
    v8f acc;
    {
      v16h ah = load_a_tile(h_s, lane, 0);
      v16h bL = load_b_tile(wl, lane, nt * 16, 0);
      acc = WMMA_F16(ah, bL, zero8);
      v16h ax = load_a_tile(xt_s, lane, 0);
      v16h bR = load_b_tile(wr, lane, nt * 16, 0);
      acc = WMMA_F16(ax, bR, acc);
    }
#pragma unroll
    for (int kt = 1; kt < 4; ++kt) {
      v16h ah = load_a_tile(h_s, lane, kt);
      v16h bL = load_b_tile(wl, lane, nt * 16, kt);
      acc = WMMA_F16(ah, bL, acc);
      v16h ax = load_a_tile(xt_s, lane, kt);
      v16h bR = load_b_tile(wr, lane, nt * 16, kt);
      acc = WMMA_F16(ax, bR, acc);
    }
    const int ncol = nt * 16 + (lane & 15);
    const int mb   = (lane >> 4) * 8;
#pragma unroll
    for (int r = 0; r < 8; ++r) {
      int node = base + mb + r;
      if (node < n_nodes) {
        float v = acc[r] + blv;
        if (RELU) v = v > 0.0f ? v : 0.0f;
        if (FINAL) out_f32[(size_t)node * FO + ncol] = v;
        else       feat_out[(size_t)node * FO + ncol] = (_Float16)v;
      }
    }
  }
}

// ---------- host side ----------
extern "C" void kernel_launch(void* const* d_in, const int* in_sizes, int n_in,
                              void* d_out, int out_size, void* d_ws, size_t ws_size,
                              hipStream_t stream) {
  (void)n_in; (void)out_size; (void)ws_size;
  const int n = in_sizes[0] / FI;             // 50000

  const float* x   = (const float*)d_in[0];
  const int*   src = (const int*)d_in[1];     // edge_index row 0 = src, [N*16]

  struct Lay { const float *wih, *whh, *bih, *bhh, *wl, *bl, *wr; int fo; };
  Lay L[3] = {
    { (const float*)d_in[2],  (const float*)d_in[3],  (const float*)d_in[4],
      (const float*)d_in[5],  (const float*)d_in[6],  (const float*)d_in[7],
      (const float*)d_in[8],  128 },
    { (const float*)d_in[9],  (const float*)d_in[10], (const float*)d_in[11],
      (const float*)d_in[12], (const float*)d_in[13], (const float*)d_in[14],
      (const float*)d_in[15], 128 },
    { (const float*)d_in[16], (const float*)d_in[17], (const float*)d_in[18],
      (const float*)d_in[19], (const float*)d_in[20], (const float*)d_in[21],
      (const float*)d_in[22], 64 },
  };

  char* p = (char*)d_ws;
  auto carve = [&](size_t bytes) -> char* {
    char* r = p; p += (bytes + 255) & ~(size_t)255; return r;
  };
  _Float16* featA = (_Float16*)carve((size_t)n * FI * 2);
  _Float16* featB = (_Float16*)carve((size_t)n * FI * 2);
  _Float16 *wihH[3], *whhH[3], *wlH[3], *wrH[3];
  float* biasS[3];
  for (int l = 0; l < 3; ++l) {
    wihH[l]  = (_Float16*)carve((size_t)512 * 128 * 2);
    whhH[l]  = (_Float16*)carve((size_t)512 * 128 * 2);
    wlH[l]   = (_Float16*)carve((size_t)L[l].fo * 128 * 2);
    wrH[l]   = (_Float16*)carve((size_t)L[l].fo * 128 * 2);
    biasS[l] = (float*)carve((size_t)512 * 4);
  }

  auto cvt = [&](const float* s, _Float16* d, int cnt) {
    int blocks = (cnt + 255) / 256; if (blocks > 16384) blocks = 16384;
    k_f32_to_f16<<<blocks, 256, 0, stream>>>(s, d, cnt);
  };

  cvt(x, featA, n * FI);
  for (int l = 0; l < 3; ++l) {
    cvt(L[l].wih, wihH[l], 512 * 128);
    cvt(L[l].whh, whhH[l], 512 * 128);
    cvt(L[l].wl,  wlH[l],  L[l].fo * 128);
    cvt(L[l].wr,  wrH[l],  L[l].fo * 128);
    k_bias_sum<<<2, 256, 0, stream>>>(L[l].bih, L[l].bhh, biasS[l], 512);
  }

  const int grid = (n + 15) / 16;  // 3125

  sage_layer<128, true, false><<<grid, 256, SAGE_SMEM, stream>>>(
      featA, src, wihH[0], whhH[0], biasS[0], wlH[0], L[0].bl, wrH[0],
      featB, (float*)nullptr, n);

  sage_layer<128, true, false><<<grid, 256, SAGE_SMEM, stream>>>(
      featB, src, wihH[1], whhH[1], biasS[1], wlH[1], L[1].bl, wrH[1],
      featA, (float*)nullptr, n);

  sage_layer<64, false, true><<<grid, 256, SAGE_SMEM, stream>>>(
      featA, src, wihH[2], whhH[2], biasS[2], wlH[2], L[2].bl, wrH[2],
      (_Float16*)nullptr, (float*)d_out, n);
}
